// LigerLMHeadDPO_77429670412358
// MI455X (gfx1250) — matrix operand order repaired and checked by reading, not compiled, not run on hardware
//
#include <hip/hip_runtime.h>
#include <hip/hip_bf16.h>
#include <stdint.h>

#define IGNORE_INDEX (-100LL)
#define BETA 0.1f

#define Bsz 4
#define T   1024
#define H   2048            // K dimension
#define Vv  32000           // vocab
#define M_TOT (Bsz * T)     // 4096 rows
#define MT  64              // rows per workgroup (x-tile in LDS = 64*2048*2 = 256KB)
#define NVB (Vv / 64)       // 500 blocks of 64 columns (4 WMMA tiles)

typedef __attribute__((ext_vector_type(16))) __bf16 v16bf;
typedef __attribute__((ext_vector_type(8)))  float  v8f;

union AFrag {
    v16bf  v;
    float4 f4[2];
};

__device__ inline unsigned short f2bf_rne(float f) {
    unsigned int u = __float_as_uint(f);
    unsigned int r = u + 0x7fffu + ((u >> 16) & 1u);
    return (unsigned short)(r >> 16);
}

#define WMMA_BF16(a, b, c) \
    __builtin_amdgcn_wmma_f32_16x16x32_bf16(false, (a), false, (b), (short)0, (c), false, false)

// ---------------------------------------------------------------------------
// Kernel 1: convert W (65,536,000 f32) and x (8,388,608 f32) to bf16 in ws.
// bf16 W = 131 MB -> fits in the 192 MB L2, making later W sweeps L2-resident.
// ---------------------------------------------------------------------------
__global__ void __launch_bounds__(256)
convert_bf16_kernel(const float* __restrict__ Wf, const float* __restrict__ Xf,
                    unsigned short* __restrict__ Wb, unsigned short* __restrict__ Xb) {
    const long long NW = (long long)Vv * H;
    const long long NX = (long long)M_TOT * H;
    long long base = ((long long)blockIdx.x * blockDim.x + threadIdx.x) * 4;
    if (base < NW) {
        float4 w = *(const float4*)(Wf + base);
        ushort4 o;
        o.x = f2bf_rne(w.x); o.y = f2bf_rne(w.y);
        o.z = f2bf_rne(w.z); o.w = f2bf_rne(w.w);
        *(ushort4*)(Wb + base) = o;
    }
    if (base < NX) {
        float4 x = *(const float4*)(Xf + base);
        ushort4 o;
        o.x = f2bf_rne(x.x); o.y = f2bf_rne(x.y);
        o.z = f2bf_rne(x.z); o.w = f2bf_rne(x.w);
        *(ushort4*)(Xb + base) = o;
    }
}

// ---------------------------------------------------------------------------
// Kernel 2: fused X @ W^T with online logsumexp per row.
// Grid: M_TOT/MT = 64 WGs, 256 threads (8 wave32).
// wave w: m-subtile = w&3 (16 rows), v-half = w>>2.
// Each wave computes a 16x64 output block per iteration: 4 independent
// accumulators -> 4 parallel WMMA chains, A-fragment reused 4x.
// Online logsumexp is PER-LANE in the hot loop (no cross-lane traffic);
// lanes are merged once at the end.
// ---------------------------------------------------------------------------
__global__ void __launch_bounds__(256)
dpo_lse_kernel(const unsigned short* __restrict__ Wb,
               const unsigned short* __restrict__ Xb,
               float* __restrict__ lse_out) {
    extern __shared__ char smem[];
    unsigned short* xs = (unsigned short*)smem;                 // MT*H bf16
    float* mbuf = (float*)(smem + (size_t)MT * H * 2);          // [2][MT]
    float* sbuf = mbuf + 2 * MT;                                // [2][MT]

    const int wg   = blockIdx.x;
    const int row0 = wg * MT;
    const int tid  = threadIdx.x;

    // Stage x-tile into LDS via CDNA5 async global->LDS copies (ASYNCcnt).
    {
        const unsigned short* xsrc = Xb + (size_t)row0 * H;
        for (int i = tid * 8; i < MT * H; i += 256 * 8) {
            unsigned lds_addr = (unsigned)(uintptr_t)(xs + i);   // AS(3) byte offset
            unsigned goff     = (unsigned)(i * 2);               // byte offset
            asm volatile("global_load_async_to_lds_b128 %0, %1, %2"
                         :: "v"(lds_addr), "v"(goff), "s"(xsrc) : "memory");
        }
        asm volatile("s_wait_asynccnt 0x0" ::: "memory");
    }
    __syncthreads();

    const int wave  = tid >> 5;
    const int lane  = tid & 31;
    const int msub  = wave & 3;   // which 16-row subtile
    const int vhalf = wave >> 2;  // even / odd 64-wide V blocks
    const int n     = lane & 15;  // A: M index; B: N index
    const int half  = lane >> 4;  // half-wave selector (K split)

    const unsigned short* xrow = xs + (size_t)(msub * 16 + n) * H;

    // Per-lane online-softmax state: row (r + half*8), this lane's columns.
    float rm[8], rs[8];
#pragma unroll
    for (int r = 0; r < 8; ++r) { rm[r] = -1.0e30f; rs[r] = 0.0f; }

    for (int vb = vhalf; vb < NVB; vb += 2) {
        const unsigned short* w0 = Wb + (size_t)(vb * 64 + n) * H + half * 16;
        const unsigned short* w1 = w0 + 16 * H;
        const unsigned short* w2 = w0 + 32 * H;
        const unsigned short* w3 = w0 + 48 * H;

        v8f c0 = {0.f,0.f,0.f,0.f,0.f,0.f,0.f,0.f};
        v8f c1 = c0, c2 = c0, c3 = c0;

#pragma unroll 2
        for (int k = 0; k < H; k += 32) {
            AFrag a, b0, b1, b2, b3;
            // A 16x32 bf16: lanes 0-15 K={k..k+15}, lanes 16-31 K={k+16..k+31}
            a.f4[0] = *(const float4*)(xrow + k + half * 16);
            a.f4[1] = *(const float4*)(xrow + k + half * 16 + 8);
            // B 32x16 bf16 (= W^T tiles): lane n reads its W row, K contiguous
            b0.f4[0] = *(const float4*)(w0 + k);  b0.f4[1] = *(const float4*)(w0 + k + 8);
            b1.f4[0] = *(const float4*)(w1 + k);  b1.f4[1] = *(const float4*)(w1 + k + 8);
            b2.f4[0] = *(const float4*)(w2 + k);  b2.f4[1] = *(const float4*)(w2 + k + 8);
            b3.f4[0] = *(const float4*)(w3 + k);  b3.f4[1] = *(const float4*)(w3 + k + 8);
            c0 = WMMA_BF16(a.v, b0.v, c0);
            c1 = WMMA_BF16(a.v, b1.v, c1);
            c2 = WMMA_BF16(a.v, b2.v, c2);
            c3 = WMMA_BF16(a.v, b3.v, c3);
        }

        if (vb + 2 < NVB) {  // keep next W block hot (high temporal locality)
            __builtin_prefetch(w0 + (size_t)128 * H, 0, 3);
            __builtin_prefetch(w2 + (size_t)128 * H, 0, 3);
        }

        // Per-lane online logsumexp update (no shuffles in hot path).
#pragma unroll
        for (int r = 0; r < 8; ++r) {
            float v0 = c0[r], v1 = c1[r], v2 = c2[r], v3 = c3[r];
            float tm = fmaxf(fmaxf(v0, v1), fmaxf(v2, v3));
            float mnew = fmaxf(rm[r], tm);
            float e = __expf(v0 - mnew) + __expf(v1 - mnew) +
                      __expf(v2 - mnew) + __expf(v3 - mnew);
            rs[r] = rs[r] * __expf(rm[r] - mnew) + e;
            rm[r] = mnew;
        }
    }

    // Merge (m,s) across the 16-lane group (once, outside the hot loop).
#pragma unroll
    for (int r = 0; r < 8; ++r) {
        float m = rm[r], s = rs[r];
#pragma unroll
        for (int o = 1; o <= 8; o <<= 1) {
            float mo = __shfl_xor(m, o, 32);
            float so = __shfl_xor(s, o, 32);
            float Mx = fmaxf(m, mo);
            s = s * __expf(m - Mx) + so * __expf(mo - Mx);
            m = Mx;
        }
        rm[r] = m; rs[r] = s;
    }

    // Publish wave partials: lanes 0 and 16 hold rows msub*16 + half*8 + r.
    if ((lane & 15) == 0) {
#pragma unroll
        for (int r = 0; r < 8; ++r) {
            int rloc = msub * 16 + half * 8 + r;
            mbuf[vhalf * MT + rloc] = rm[r];
            sbuf[vhalf * MT + rloc] = rs[r];
        }
    }
    __syncthreads();

    // Merge the two v-half partials per row -> lse.
    if (tid < MT) {
        float m0 = mbuf[tid],      s0 = sbuf[tid];
        float m1 = mbuf[MT + tid], s1 = sbuf[MT + tid];
        float Mx = fmaxf(m0, m1);
        float S  = s0 * __expf(m0 - Mx) + s1 * __expf(m1 - Mx);
        lse_out[row0 + tid] = Mx + __logf(S);
    }
}

// ---------------------------------------------------------------------------
// Kernel 3: target logits in full fp32 (4096 dot products of length 2048).
// ---------------------------------------------------------------------------
__global__ void __launch_bounds__(256)
target_logit_kernel(const float* __restrict__ Xf, const long long* __restrict__ Y,
                    const float* __restrict__ Wf, float* __restrict__ tgt) {
    const int row = blockIdx.x;
    const long long y = Y[row];
    float acc = 0.0f;
    if (y >= 0 && y < Vv) {
        const float* xr = Xf + (size_t)row * H;
        const float* wr = Wf + (size_t)y * H;
        for (int k = threadIdx.x * 4; k < H; k += 256 * 4) {
            float4 xv = *(const float4*)(xr + k);
            float4 wv = *(const float4*)(wr + k);
            acc += xv.x * wv.x + xv.y * wv.y + xv.z * wv.z + xv.w * wv.w;
        }
    }
    for (int o = 16; o >= 1; o >>= 1) acc += __shfl_xor(acc, o, 32);
    __shared__ float red[8];
    const int lane = threadIdx.x & 31, wave = threadIdx.x >> 5;
    if (lane == 0) red[wave] = acc;
    __syncthreads();
    if (threadIdx.x == 0) {
        float s = 0.0f;
#pragma unroll
        for (int w = 0; w < 8; ++w) s += red[w];
        tgt[row] = s;
    }
}

// ---------------------------------------------------------------------------
// Kernel 4: per-sequence masked means + DPO loss (scalar).
// ---------------------------------------------------------------------------
__global__ void __launch_bounds__(128)
dpo_loss_kernel(const long long* __restrict__ Y, const float* __restrict__ lse,
                const float* __restrict__ tgt, float* __restrict__ out) {
    const int wave = threadIdx.x >> 5, lane = threadIdx.x & 31;
    float sum = 0.0f, cnt = 0.0f;
    for (int t = lane; t < T; t += 32) {
        const int idx = wave * T + t;
        long long yv = Y[idx];
        if (yv != IGNORE_INDEX) {
            sum += tgt[idx] - lse[idx];
            cnt += 1.0f;
        }
    }
    for (int o = 16; o >= 1; o >>= 1) {
        sum += __shfl_xor(sum, o, 32);
        cnt += __shfl_xor(cnt, o, 32);
    }
    __shared__ float sseq[4];
    if (lane == 0) sseq[wave] = sum / fmaxf(cnt, 1.0f);
    __syncthreads();
    if (threadIdx.x == 0) {
        float loss = 0.0f;
#pragma unroll
        for (int p = 0; p < 2; ++p) {
            float z = BETA * (sseq[p] - sseq[p + 2]);
            float ls = fminf(z, 0.0f) - log1pf(__expf(-fabsf(z)));  // log_sigmoid
            loss -= ls;
        }
        out[0] = loss * 0.5f;  // / n_pairs
    }
}

// ---------------------------------------------------------------------------
extern "C" void kernel_launch(void* const* d_in, const int* in_sizes, int n_in,
                              void* d_out, int out_size, void* d_ws, size_t ws_size,
                              hipStream_t stream) {
    const float*     x  = (const float*)d_in[0];      // (4,1024,2048) f32
    const long long* y  = (const long long*)d_in[1];  // (4,1024) i64
    const float*     Wf = (const float*)d_in[2];      // (32000,2048) f32
    float* out = (float*)d_out;

    char* ws = (char*)d_ws;
    unsigned short* Wb = (unsigned short*)ws;                         // 131,072,000 B
    unsigned short* Xb = (unsigned short*)(ws + 131072000LL);         //  16,777,216 B
    float* lse = (float*)(ws + 131072000LL + 16777216LL);             //  16,384 B
    float* tgt = lse + M_TOT;

    {
        long long nthreads = ((long long)Vv * H) / 4;   // 16,384,000
        int blocks = (int)(nthreads / 256);             // 64,000
        convert_bf16_kernel<<<blocks, 256, 0, stream>>>(Wf, x, Wb, Xb);
    }

    {
        size_t smem = (size_t)MT * H * 2 + 2 * 2 * MT * sizeof(float); // 263,168 B
        hipFuncSetAttribute((const void*)dpo_lse_kernel,
                            hipFuncAttributeMaxDynamicSharedMemorySize, (int)smem);
        dpo_lse_kernel<<<M_TOT / MT, 256, smem, stream>>>(Wb, Xb, lse);
    }

    target_logit_kernel<<<M_TOT, 256, 0, stream>>>(x, y, Wf, tgt);

    dpo_loss_kernel<<<1, 128, 0, stream>>>(y, lse, tgt, out);
}
